// FilteredNoise_55301998903905
// MI455X (gfx1250) — matrix operand order repaired and checked by reading, not compiled, not run on hardware
//
#include <hip/hip_runtime.h>

// FilteredNoise on MI455X (gfx1250):
//   Stage A: H = Z @ W  via v_wmma_f32_16x16x32_f16 (W = folded irfft+roll+Hann basis,
//            generated on-device directly in WMMA B-operand lane layout).
//   Stage B: per-frame 256 (x) 513 convolution + overlap-add (hop 256) on fp32 VALU.
//            LDS staged with GLOBAL_LOAD_ASYNC_TO_LDS_B128 (ASYNCcnt path), inner loop
//            FMA-bound: 1 b128 broadcast per 4 FMAs + 1 b32 per-lane read per FMA.

typedef __attribute__((ext_vector_type(16))) _Float16 v16h;
typedef __attribute__((ext_vector_type(8)))  float    v8f;

#define NB     32
#define NFR    1000
#define NL     257            // rfft bins of z
#define IRLEN  513            // 2L-1
#define FLEN   256            // noise frame length
#define ROWS   (NB * NFR)     // 32000 flattened (b,fr) rows
#define KT     9              // K tiles of 32  -> KPAD = 288
#define KPAD   (KT * 32)
#define TT     33             // T tiles of 16  -> TPAD = 528
#define TPAD   (TT * 16)
#define RT     (ROWS / 16)    // 2000 row tiles
#define TG     11             // groups of 3 column tiles (33 = 11*3)
#define STILES 1002           // output tiles of 256: 999*256+768 = 1002*256
#define TWO_PI 6.28318530717958647692f

// workspace byte offsets (total ~86.5 MB)
#define WPK_OFF ((size_t)0)                                  // 288*528 f16 = 304,128 B
#define ZPK_OFF ((size_t)(512 * 1024))                       // 32000*288 f16 = 18,432,000 B
#define H_OFF   (ZPK_OFF + (size_t)ROWS * KPAD * 2)          // 32000*528 f32 = 67,584,000 B

// ---------------------------------------------------------------------------
// CDNA5 async global->LDS copy (ASYNCcnt), 16 bytes per active lane.
// VDST = VGPR holding LDS byte address, VADDR = 64-bit global address, GV mode.
__device__ __forceinline__ void async_copy_b128(void* lds_dst, const void* gsrc) {
  uint32_t l = (uint32_t)(uintptr_t)lds_dst;   // low 32 bits of flat shared ptr
  asm volatile("global_load_async_to_lds_b128 %0, %1, off"
               :: "v"(l), "v"(gsrc)
               : "memory");
}
__device__ __forceinline__ void wait_async0() {
  asm volatile("s_wait_asynccnt 0x0" ::: "memory");
}

// ---------------------------------------------------------------------------
// Build W as WMMA B operand (16-bit B 32x16, wave32), contiguous k order, same as pack_z:
//   v16h element j of lane n  ->  K = kt*32 + (n<16 ? 0 : 16) + j, col = tt*16 + (n&15)
// W[k,t] = (c_k/513) * cos(2*pi*k*(t-256)/513) * 0.5*(1 - cos(2*pi*t/513)), zero-padded.
__global__ void init_w(_Float16* __restrict__ wpk) {
  int tid = blockIdx.x * blockDim.x + threadIdx.x;
  if (tid >= KT * TT * 32) return;
  int lane  = tid & 31;
  int tile  = tid >> 5;            // tile = kt*TT + tt
  int kt    = tile / TT;
  int tt    = tile % TT;
  int col   = tt * 16 + (lane & 15);
  int kbase = kt * 32 + ((lane & 16) ? 16 : 0);
  _Float16* dst = wpk + ((size_t)tile * 32 + lane) * 16;
  float win = 0.0f;
  if (col < IRLEN) win = 0.5f * (1.0f - __cosf(TWO_PI * (float)col / 513.0f));
#pragma unroll
  for (int j = 0; j < 16; ++j) {
    int   k = kbase + j;
    float v = 0.0f;
    if (k < NL && col < IRLEN) {
      int   m  = (k * (col - 256)) % 513;      // exact integer phase reduction
      float ck = (k == 0) ? 1.0f : 2.0f;
      v = (ck / 513.0f) * __cosf(TWO_PI * (float)m / 513.0f) * win;
    }
    dst[j] = (_Float16)v;
  }
}

// ---------------------------------------------------------------------------
// Pack Z as WMMA A operand (16-bit A 16x32, wave32), contiguous k order, same as init_w:
//   lane L row = 16*rt + (L&15); element j -> K = kt*32 + (L<16?0:16) + j
__global__ void pack_z(const float* __restrict__ z, _Float16* __restrict__ zpk) {
  int tid   = blockIdx.x * blockDim.x + threadIdx.x;   // RT*KT*32 threads exactly
  int lane  = tid & 31;
  int tile  = tid >> 5;            // tile = rt*KT + kt
  int rt    = tile / KT;
  int kt    = tile % KT;
  int row   = rt * 16 + (lane & 15);
  int kbase = kt * 32 + ((lane & 16) ? 16 : 0);
  const float* src = z + (size_t)row * NL;
  _Float16*    dst = zpk + ((size_t)tile * 32 + lane) * 16;
#pragma unroll
  for (int j = 0; j < 16; ++j) {
    int k = kbase + j;
    dst[j] = (_Float16)((k < NL) ? src[k] : 0.0f);
  }
}

// ---------------------------------------------------------------------------
// H(32000 x 528) = Zpk(32000 x 288) @ Wpk(288 x 528), fp16 in / fp32 accumulate.
// One wave per (row-tile, group of 3 column-tiles): 9 K-steps, 27 WMMAs, A reused 3x.
__global__ __launch_bounds__(256) void gemm_h(const v16h* __restrict__ zpk,
                                              const v16h* __restrict__ wpk,
                                              float* __restrict__ H) {
  int gtid = blockIdx.x * blockDim.x + threadIdx.x;
  int wave = gtid >> 5;                 // 0 .. RT*TG-1 (exact: 22000 = 2750*8)
  int lane = threadIdx.x & 31;
  int rt   = wave / TG;
  int tg   = wave % TG;
  v8f acc0 = {}; v8f acc1 = {}; v8f acc2 = {};
#pragma unroll
  for (int kt = 0; kt < KT; ++kt) {
    // prefetch next streamed A k-tile (B stays hot in L2: only 304 KB total)
    int ktn = (kt < KT - 1) ? kt + 1 : kt;
    __builtin_prefetch(&zpk[((size_t)rt * KT + ktn) * 32 + lane], 0, 3);
    v16h a  = zpk[((size_t)rt * KT + kt) * 32 + lane];
    v16h b0 = wpk[((size_t)kt * TT + tg * 3 + 0) * 32 + lane];
    v16h b1 = wpk[((size_t)kt * TT + tg * 3 + 1) * 32 + lane];
    v16h b2 = wpk[((size_t)kt * TT + tg * 3 + 2) * 32 + lane];
    acc0 = __builtin_amdgcn_wmma_f32_16x16x32_f16(false, a, false, b0, (short)0, acc0, false, false);
    acc1 = __builtin_amdgcn_wmma_f32_16x16x32_f16(false, a, false, b1, (short)0, acc1, false, false);
    acc2 = __builtin_amdgcn_wmma_f32_16x16x32_f16(false, a, false, b2, (short)0, acc2, false, false);
  }
  // C/D layout: VGPR v -> M = v + (lane<16 ? 0 : 8), N = lane&15
  int mbase = rt * 16 + ((lane & 16) ? 8 : 0);
  int nbase = lane & 15;
#pragma unroll
  for (int v = 0; v < 8; ++v) {
    size_t rowoff = (size_t)(mbase + v) * TPAD;
    H[rowoff + (size_t)(tg * 3 + 0) * 16 + nbase] = acc0[v];
    H[rowoff + (size_t)(tg * 3 + 1) * 16 + nbase] = acc1[v];
    H[rowoff + (size_t)(tg * 3 + 2) * 16 + nbase] = acc2[v];
  }
}

// ---------------------------------------------------------------------------
// conv + overlap-add. Block = (b, s): 256 outputs out[b, s*256 + j].
// out = 0.01 * sum_{d=0..2} sum_tau noise[b, s-d, tau] * H[b, s-d, d*256 + j - tau]
// H staged zero-padded in LDS via async b128 copies; inner loops are branch-free FMAs:
// nz read as aligned float4 broadcast, hp[... - tau] lane-consecutive (bank-clean).
__global__ __launch_bounds__(256) void conv_ola(const float* __restrict__ noise,
                                                const float* __restrict__ H,
                                                float* __restrict__ out) {
  __shared__ __align__(16) float nz[3][FLEN];
  __shared__ __align__(16) float hp[3][1024];  // [0..255]=0 | [256..783]=H row (528) | [784..1023]=0
  int b = blockIdx.x, s = blockIdx.y, tid = threadIdx.x;
  for (int d = 0; d < 3; ++d) {
    int  fr = s - d;
    bool ok = (fr >= 0) && (fr < NFR);
    size_t r = (size_t)b * NFR + (ok ? fr : 0);
    hp[d][tid] = 0.0f;                          // front pad [0..255]
    if (tid < 240) hp[d][784 + tid] = 0.0f;     // tail pad [784..1023]
    if (ok) {
      // async global->LDS, 16B/lane: H padded row (528 floats = 132 x b128, H[513..527]==0)
      if (tid < 132) async_copy_b128(&hp[d][256 + tid * 4], &H[r * TPAD + tid * 4]);
      if (tid < 64)  async_copy_b128(&nz[d][tid * 4], &noise[r * FLEN + tid * 4]);
    } else {
      for (int t = tid; t < 528; t += 256) hp[d][256 + t] = 0.0f;
      nz[d][tid] = 0.0f;
    }
  }
  wait_async0();          // our wave's async LDS writes done
  __syncthreads();        // everyone's writes visible
  float acc = 0.0f;
#pragma unroll
  for (int d = 0; d < 3; ++d) {
    const float* h = &hp[d][256 + d * 256 + tid];
#pragma unroll 4
    for (int tau = 0; tau < FLEN; tau += 4) {
      float4 nv = *(const float4*)&nz[d][tau];  // one ds_load_b128 broadcast / 4 FMAs
      acc = __builtin_fmaf(nv.x, h[-tau - 0], acc);
      acc = __builtin_fmaf(nv.y, h[-tau - 1], acc);
      acc = __builtin_fmaf(nv.z, h[-tau - 2], acc);
      acc = __builtin_fmaf(nv.w, h[-tau - 3], acc);
    }
  }
  out[((size_t)b * STILES + s) * FLEN + tid] = 0.01f * acc;
}

// ---------------------------------------------------------------------------
extern "C" void kernel_launch(void* const* d_in, const int* in_sizes, int n_in,
                              void* d_out, int out_size, void* d_ws, size_t ws_size,
                              hipStream_t stream) {
  const float* z     = (const float*)d_in[0];   // (32,1000,257) f32
  const float* noise = (const float*)d_in[1];   // (32,1000,256) f32
  float*       out   = (float*)d_out;           // (32,256512)   f32
  char*        ws    = (char*)d_ws;             // needs ~86.5 MB
  _Float16* wpk = (_Float16*)(ws + WPK_OFF);
  _Float16* zpk = (_Float16*)(ws + ZPK_OFF);
  float*    H   = (float*)   (ws + H_OFF);

  init_w<<<(KT * TT * 32 + 255) / 256, 256, 0, stream>>>(wpk);
  pack_z<<<(RT * KT * 32) / 256, 256, 0, stream>>>(z, zpk);          // 2250 blocks exact
  gemm_h<<<(RT * TG) / 8, 256, 0, stream>>>((const v16h*)zpk, (const v16h*)wpk, H);
  dim3 g(NB, STILES);
  conv_ola<<<g, 256, 0, stream>>>(noise, H, out);
}
